// RegionAdaTransform_7026566496838
// MI455X (gfx1250) — matrix operand order repaired and checked by reading, not compiled, not run on hardware
//
#include <hip/hip_runtime.h>
#include <hip/hip_bf16.h>

// ---------------------------------------------------------------------------
// Types for CDNA5 WMMA (wave32): bf16 16x16x32, f32 accumulate
// ---------------------------------------------------------------------------
typedef __attribute__((ext_vector_type(16))) __bf16 v16bf;
typedef __attribute__((ext_vector_type(8)))  float  v8f;

union BF16x16 { v16bf v; uint4 q[2]; };

__device__ __forceinline__ unsigned short f2bf(float f) {
    unsigned u = __float_as_uint(f);
    unsigned r = (u + 0x7FFFu + ((u >> 16) & 1u)) >> 16;   // round-to-nearest-even
    return (unsigned short)r;
}
__device__ __forceinline__ float bf2f(unsigned short h) {
    return __uint_as_float(((unsigned)h) << 16);
}

__device__ __forceinline__ v8f wmma_bf16(v16bf a, v16bf b, v8f c) {
    return __builtin_amdgcn_wmma_f32_16x16x32_bf16(
        false, a, false, b, (short)0, c, false, false);
}

// A fragment (16x32 bf16, M x K), row-major source with leading dim `ld` halves.
// ISA layout: lane<16 -> M=lane, elems 0..7 = K 0..7, elems 8..15 = K 16..23;
//             lane>=16 -> M=lane-16, K offsets +8.  Two 16-byte loads per lane.
__device__ __forceinline__ v16bf load_a_frag(const unsigned short* A, int ld,
                                             int rowBase, int kBase) {
    int lane = threadIdx.x & 31;
    const unsigned short* p =
        A + (size_t)(rowBase + (lane & 15)) * ld + kBase + ((lane >> 4) << 3);
    BF16x16 r;
    r.q[0] = *(const uint4*)p;          // K = koff .. koff+7
    r.q[1] = *(const uint4*)(p + 16);   // K = koff+16 .. koff+23
    return r.v;
}

// B fragment (32x16 bf16, K x N) from pre-swizzled weights:
// storage index = (((kt*12 + nt)*32) + lane)*16 + e, where
// K = kt*32 + (lane>=16?8:0) + (e&7) + (e>=8?16:0), N = nt*16 + (lane&15). 12 N-tiles (N=192).
__device__ __forceinline__ v16bf load_b_frag(const unsigned short* Bswz,
                                             int kt, int nt) {
    int lane = threadIdx.x & 31;
    const unsigned short* p = Bswz + (((size_t)(kt * 12 + nt) * 32) + lane) * 16;
    BF16x16 r;
    r.q[0] = *(const uint4*)p;
    r.q[1] = *(const uint4*)(p + 8);
    return r.v;
}

// ---------------------------------------------------------------------------
// Prep kernels
// ---------------------------------------------------------------------------
__global__ void fill_zero_kernel(unsigned int* p, size_t n) {
    size_t i = (size_t)blockIdx.x * blockDim.x + threadIdx.x;
    size_t stride = (size_t)gridDim.x * blockDim.x;
    for (; i < n; i += stride) p[i] = 0u;
}

// W is (O=192, Ktot) fp32 (1x1 conv weight). Emit B-matrix [k=i][n=o] swizzled.
__global__ void swizzleB_kernel(const float* __restrict__ W, int Ktot, int iOff,
                                unsigned short* __restrict__ out) {
    int tid = blockIdx.x * 256 + threadIdx.x;
    if (tid >= 6 * 12 * 32 * 16) return;
    int e = tid & 15, lane = (tid >> 4) & 31;
    int rem = tid >> 9; int nt = rem % 12, kt = rem / 12;
    int k = kt * 32 + ((lane >> 4) << 3) + (e & 7) + ((e >> 3) << 4);
    int n = nt * 16 + (lane & 15);
    out[tid] = f2bf(W[(size_t)n * Ktot + iOff + k]);
}

// ws1 (192,192,3,3): per-tap swizzled B matrices, tap t = di*3+dj.
__global__ void swizzleB_conv3_kernel(const float* __restrict__ W,
                                      unsigned short* __restrict__ out) {
    const int TAPSZ = 6 * 12 * 32 * 16;      // 36864
    int tid = blockIdx.x * 256 + threadIdx.x;
    if (tid >= 9 * TAPSZ) return;
    int t = tid / TAPSZ;
    int r = tid % TAPSZ;
    int e = r & 15, lane = (r >> 4) & 31;
    int rem = r >> 9; int nt = rem % 12, kt = rem / 12;
    int k = kt * 32 + ((lane >> 4) << 3) + (e & 7) + ((e >> 3) << 4);
    int n = nt * 16 + (lane & 15);
    out[tid] = f2bf(W[((size_t)(n * 192 + k)) * 9 + t]);
}

// NCHW fp32 input -> NHWC bf16 A-matrix; gather mask centers -> NHWC bf16.
__global__ void prep_acts_kernel(const float* __restrict__ input,
                                 const float* __restrict__ centers,
                                 const int* __restrict__ mask,
                                 unsigned short* __restrict__ X0bf,
                                 unsigned short* __restrict__ CEbf) {
    int tid = blockIdx.x * 256 + threadIdx.x;
    if (tid >= 32768 * 192) return;
    int c = tid % 192; int row = tid / 192;
    int b = row >> 12; int p = row & 4095;
    X0bf[tid] = f2bf(input[(((size_t)b * 192 + c) << 12) + p]);
    int m = mask[(b << 12) + p];
    CEbf[tid] = f2bf(centers[((size_t)(b * 16 + m)) * 192 + c]);
}

// ---------------------------------------------------------------------------
// GEMM: C[P,192] = act(A[P,K] @ B[K,192] + bias). KT2 = K/32 (6 or 12=concat).
// Block = 256 threads = 8 waves; each wave owns a 16-row stripe, loops 12
// N-tiles with A fragments held in registers. launch_bounds(256,1) lifts the
// VGPR cap so the 12 A-fragments of the concat variant stay in registers.
// ---------------------------------------------------------------------------
template <int KT2>
__global__ void __launch_bounds__(256, 1)
gemm192_kernel(const unsigned short* __restrict__ A,
               const unsigned short* __restrict__ A2,
               const unsigned short* __restrict__ Bswz,
               const unsigned short* __restrict__ B2swz,
               const float* __restrict__ bias, int act,
               unsigned short* __restrict__ outA,
               unsigned short* __restrict__ outPadNHWC,
               unsigned short* __restrict__ outPadNCHW) {
    const int lane = threadIdx.x & 31;
    const int wave = threadIdx.x >> 5;
    const int rowBase = blockIdx.x * 128 + wave * 16;

    v16bf af[KT2];
#pragma unroll
    for (int kt = 0; kt < KT2; ++kt) {
        if (kt < 6) af[kt] = load_a_frag(A, 192, rowBase, kt * 32);
        else        af[kt] = load_a_frag(A2, 192, rowBase, (kt - 6) * 32);
    }
    for (int nt = 0; nt < 12; ++nt) {
        float bv = bias[nt * 16 + (lane & 15)];
        v8f acc = {bv, bv, bv, bv, bv, bv, bv, bv};
#pragma unroll
        for (int kt = 0; kt < KT2; ++kt) {
            const unsigned short* bs = (kt < 6) ? Bswz : B2swz;
            int ktl = (kt < 6) ? kt : kt - 6;
            acc = wmma_bf16(af[kt], load_b_frag(bs, ktl, nt), acc);
        }
#pragma unroll
        for (int v = 0; v < 8; ++v) {
            float x = acc[v];
            if (act) x = (x < 0.f) ? 0.1f * x : x;
            unsigned short hb = f2bf(x);
            int row = rowBase + v + ((lane >> 4) << 3);
            int ch = nt * 16 + (lane & 15);
            int b = row >> 12, p = row & 4095, hh = p >> 6, ww = p & 63;
            if (outA)
                outA[(size_t)row * 192 + ch] = hb;
            if (outPadNHWC)   // [b][66][66][192] zero-padded
                outPadNHWC[(((size_t)(b * 66 + hh + 1)) * 66 + ww + 1) * 192 + ch] = hb;
            if (outPadNCHW)   // [b][192][66][66] zero-padded
                outPadNCHW[(((size_t)(b * 192 + ch)) * 66 + hh + 1) * 66 + ww + 1] = hb;
        }
    }
}

// ---------------------------------------------------------------------------
// 3x3 conv as implicit GEMM over padded NHWC bf16, K = 9 taps * 192.
// Block = 256 threads = 8 waves, covers TWO image rows (128 px = 8 M-tiles x
// 12 N-tiles). The 4-row x 66-col x 192-ch halo tile (~101 KB) is staged into
// LDS once per block (CDNA5 WGP has 320 KB); A fragments then come from
// ds_load_b128. LDS pixel pitch padded 192->200 halves => the 16 lanes of an
// A-fragment start at stride 400 B = 36 banks (mod 64), all distinct => no
// bank conflicts. Each wave: 4 M-tiles x 3 N-tiles => every B fragment fetched
// from L2 feeds 4 WMMAs. Inner loop keeps 4 A-fragments + ONE B-fragment live
// (12 acc = 96 VGPR + 4 A = 32 + 1 B = 8); launch_bounds(256,1) lifts the
// VGPR cap so nothing spills to scratch.
// ---------------------------------------------------------------------------
#define C3_PITCH 200          // halves per pixel in LDS (192 + 8 pad)
#define C3_U4PP  25           // uint4 per pixel slot (24 data + 1 pad)

__device__ __forceinline__ v16bf lds_a_frag(const unsigned short* lds,
                                            int lrow, int col, int kBase) {
    int lane = threadIdx.x & 31;
    const unsigned short* p =
        lds + (size_t)(lrow * 66 + col + (lane & 15)) * C3_PITCH +
        kBase + ((lane >> 4) << 3);
    BF16x16 r;
    r.q[0] = *(const uint4*)p;
    r.q[1] = *(const uint4*)(p + 16);
    return r.v;
}

__global__ void __launch_bounds__(256, 1)
conv3x3_kernel(const unsigned short* __restrict__ Y0p,
               const unsigned short* __restrict__ Wswz9,
               const float* __restrict__ bias,
               unsigned short* __restrict__ Y1p) {
    extern __shared__ unsigned short ldsA[];        // 4*66*200 halves
    const int TAPSZ = 6 * 12 * 32 * 16;
    const int lane = threadIdx.x & 31;
    const int wave = threadIdx.x >> 5;
    const int mtg = wave & 1;                       // which of the 2 rows
    const int ntg = wave >> 1;                      // 0..3 -> nt = ntg*3+{0,1,2}
    const int b = blockIdx.x >> 5;                  // 8 batches
    const int h0 = (blockIdx.x & 31) * 2;           // row pair base

    __builtin_prefetch(Wswz9, 0, 1);                // global_prefetch_b8

    // --- cooperative stage: padded rows h0..h0+3, full 66 cols, 192 ch ------
    {
        const uint4* src =
            (const uint4*)(Y0p + ((size_t)(b * 66 + h0)) * 66 * 192);
        uint4* dst = (uint4*)ldsA;
        // 264 pixels * 24 uint4 of data each; LDS slot pitch = 25 uint4
        for (int idx = threadIdx.x; idx < 264 * 24; idx += 256) {
            int pix = idx / 24, part = idx - pix * 24;
            dst[pix * C3_U4PP + part] = src[pix * 24 + part];
        }
    }
    __syncthreads();

    // --- accumulators: 4 M-tiles (col quarters of row h0+mtg) x 3 N-tiles ---
    v8f acc[4][3];
#pragma unroll
    for (int q = 0; q < 4; ++q)
#pragma unroll
        for (int j = 0; j < 3; ++j) {
            float bv = bias[(ntg * 3 + j) * 16 + (lane & 15)];
            acc[q][j] = (v8f){bv, bv, bv, bv, bv, bv, bv, bv};
        }

    for (int t = 0; t < 9; ++t) {
        int di = t / 3, dj = t % 3;
        const unsigned short* Wt = Wswz9 + (size_t)t * TAPSZ;
        int lrow = mtg + di;
#pragma unroll
        for (int kt = 0; kt < 6; ++kt) {
            v16bf a[4];
#pragma unroll
            for (int q = 0; q < 4; ++q)
                a[q] = lds_a_frag(ldsA, lrow, q * 16 + dj, kt * 32);
#pragma unroll
            for (int j = 0; j < 3; ++j) {
                v16bf bf = load_b_frag(Wt, kt, ntg * 3 + j);
#pragma unroll
                for (int q = 0; q < 4; ++q)
                    acc[q][j] = wmma_bf16(a[q], bf, acc[q][j]);
            }
        }
    }

    // --- lrelu + write NCHW padded for the fused dynamic-conv kernel --------
    const int hh = h0 + mtg;
#pragma unroll
    for (int q = 0; q < 4; ++q)
#pragma unroll
        for (int j = 0; j < 3; ++j) {
            int ch = (ntg * 3 + j) * 16 + (lane & 15);
            unsigned short* dst =
                Y1p + (((size_t)(b * 192 + ch)) * 66 + hh + 1) * 66 + 1;
#pragma unroll
            for (int v = 0; v < 8; ++v) {
                int m = v + ((lane >> 4) << 3);
                float x = acc[q][j][v];
                x = (x < 0.f) ? 0.1f * x : x;
                dst[q * 16 + m] = f2bf(x);
            }
        }
}

// ---------------------------------------------------------------------------
// Fused depthwise kernel-gen + dynamic conv (avoids 226 MB k_sp tensor):
// out[b,c,h,w] = sum_t (bs2[c*9+t] + sum_s y1p[s]*ws2[c*9+t,s]) * x2p[t]
// One block per (b,c); per-channel weights staged in LDS.
// ---------------------------------------------------------------------------
__global__ void kdyn_kernel(const unsigned short* __restrict__ Y1p,
                            const unsigned short* __restrict__ X2p,
                            const float* __restrict__ ws2,
                            const float* __restrict__ bs2,
                            float* __restrict__ OUT0) {
    int bc = blockIdx.x;                 // 0..1535 = b*192 + c
    int c = bc % 192;
    __shared__ float w2[81];
    __shared__ float b2[9];
    if (threadIdx.x < 81) w2[threadIdx.x] = ws2[(size_t)c * 81 + threadIdx.x];
    if (threadIdx.x < 9)  b2[threadIdx.x] = bs2[c * 9 + threadIdx.x];
    __syncthreads();
    const unsigned short* y1c = Y1p + (size_t)bc * 66 * 66;
    const unsigned short* x2c = X2p + (size_t)bc * 66 * 66;
    float* outc = OUT0 + (size_t)bc * 4096;
    for (int p = threadIdx.x; p < 4096; p += 256) {
        int hh = p >> 6, ww = p & 63;
        float y[9], x[9];
#pragma unroll
        for (int s = 0; s < 9; ++s) {
            size_t off = (size_t)(hh + s / 3) * 66 + ww + (s % 3);
            y[s] = bf2f(y1c[off]);
            x[s] = bf2f(x2c[off]);
        }
        float o = 0.f;
#pragma unroll
        for (int t = 0; t < 9; ++t) {
            float ks = b2[t];
#pragma unroll
            for (int s = 0; s < 9; ++s) ks = fmaf(y[s], w2[t * 9 + s], ks);
            o = fmaf(ks, x[t], o);
        }
        outc[p] = o;
    }
}

// ---------------------------------------------------------------------------
// Squeeze-excite
// ---------------------------------------------------------------------------
__global__ void gap_reduce_kernel(const float* __restrict__ OUT0,
                                  float* __restrict__ g0) {
    int bc = blockIdx.x;
    const float* p = OUT0 + (size_t)bc * 4096;
    float s = 0.f;
    for (int i = threadIdx.x; i < 4096; i += 256) s += p[i];
    __shared__ float sm[256];
    sm[threadIdx.x] = s; __syncthreads();
    for (int off = 128; off > 0; off >>= 1) {
        if (threadIdx.x < off) sm[threadIdx.x] += sm[threadIdx.x + off];
        __syncthreads();
    }
    if (threadIdx.x == 0) g0[bc] = sm[0] * (1.0f / 4096.0f);
}

__global__ void se_gate_kernel(const float* __restrict__ g0,
                               const float* __restrict__ wc0,
                               const float* __restrict__ bc0,
                               const float* __restrict__ wc1,
                               const float* __restrict__ bc1,
                               float* __restrict__ gate) {
    int b = blockIdx.x; int o = threadIdx.x;   // 192 threads
    __shared__ float gv[192], tv[192];
    gv[o] = g0[b * 192 + o];
    __syncthreads();
    float a = bc0[o];
    for (int i = 0; i < 192; ++i) a = fmaf(wc0[(size_t)o * 192 + i], gv[i], a);
    tv[o] = (a < 0.f) ? 0.1f * a : a;
    __syncthreads();
    float c2 = bc1[o];
    for (int i = 0; i < 192; ++i) c2 = fmaf(wc1[(size_t)o * 192 + i], tv[i], c2);
    gate[b * 192 + o] = 1.0f / (1.0f + __expf(-c2));
}

__global__ void gate_apply_kernel(const float* __restrict__ OUT0,
                                  const float* __restrict__ gate,
                                  unsigned short* __restrict__ OUT1bf) {
    int i = blockIdx.x * 256 + threadIdx.x;   // over NCHW fp32
    if (i >= 32768 * 192) return;
    int p = i & 4095; int c = (i >> 12) % 192; int b = i / (192 * 4096);
    float v = OUT0[i] * gate[b * 192 + c];
    int row = b * 4096 + p;
    OUT1bf[(size_t)row * 192 + c] = f2bf(v);  // NHWC bf16 A-layout
}

// ---------------------------------------------------------------------------
// Final 1x1 conv + bias + residual, write fp32 NCHW to d_out
// ---------------------------------------------------------------------------
__global__ void __launch_bounds__(256, 1)
gemm_out_kernel(const unsigned short* __restrict__ A,
                const unsigned short* __restrict__ Bswz,
                const float* __restrict__ bias,
                const float* __restrict__ resid,
                float* __restrict__ out) {
    const int lane = threadIdx.x & 31;
    const int wave = threadIdx.x >> 5;
    const int rowBase = blockIdx.x * 128 + wave * 16;
    v16bf af[6];
#pragma unroll
    for (int kt = 0; kt < 6; ++kt) af[kt] = load_a_frag(A, 192, rowBase, kt * 32);
    for (int nt = 0; nt < 12; ++nt) {
        float bv = bias[nt * 16 + (lane & 15)];
        v8f acc = {bv, bv, bv, bv, bv, bv, bv, bv};
#pragma unroll
        for (int kt = 0; kt < 6; ++kt)
            acc = wmma_bf16(af[kt], load_b_frag(Bswz, kt, nt), acc);
#pragma unroll
        for (int v = 0; v < 8; ++v) {
            int row = rowBase + v + ((lane >> 4) << 3);
            int ch = nt * 16 + (lane & 15);
            int b = row >> 12, p = row & 4095;
            size_t idx = ((size_t)(b * 192 + ch) << 12) + p;
            out[idx] = acc[v] + resid[idx];
        }
    }
}

// ---------------------------------------------------------------------------
// Launch
// ---------------------------------------------------------------------------
extern "C" void kernel_launch(void* const* d_in, const int* in_sizes, int n_in,
                              void* d_out, int out_size, void* d_ws, size_t ws_size,
                              hipStream_t stream) {
    const float* input   = (const float*)d_in[0];
    const float* centers = (const float*)d_in[1];
    const int*   mask    = (const int*)d_in[2];
    const float* w0  = (const float*)d_in[3];
    const float* b0  = (const float*)d_in[4];
    const float* w1  = (const float*)d_in[5];
    const float* b1  = (const float*)d_in[6];
    const float* ws0 = (const float*)d_in[7];
    const float* bs0 = (const float*)d_in[8];
    const float* ws1 = (const float*)d_in[9];
    const float* bs1 = (const float*)d_in[10];
    const float* ws2 = (const float*)d_in[11];
    const float* bs2 = (const float*)d_in[12];
    const float* wc0 = (const float*)d_in[13];
    const float* bc0 = (const float*)d_in[14];
    const float* wc1 = (const float*)d_in[15];
    const float* bc1 = (const float*)d_in[16];
    const float* wo  = (const float*)d_in[17];
    const float* bo  = (const float*)d_in[18];
    float* out = (float*)d_out;

    const size_t P = 32768;                  // 8 * 64 * 64
    const size_t PADN = (size_t)8 * 192 * 66 * 66;

    char* ws = (char*)d_ws;
    size_t off = 0;
    auto alloc = [&](size_t bytes) -> void* {
        void* p = ws + off;
        off = (off + bytes + 255) & ~(size_t)255;
        return p;
    };

    unsigned short* X0bf   = (unsigned short*)alloc(P * 192 * 2);
    unsigned short* CEbf   = (unsigned short*)alloc(P * 192 * 2);
    unsigned short* X1bf   = (unsigned short*)alloc(P * 192 * 2);
    unsigned short* X2bf   = (unsigned short*)alloc(P * 192 * 2);
    unsigned short* OUT1bf = (unsigned short*)alloc(P * 192 * 2);
    unsigned short* X2p    = (unsigned short*)alloc(PADN * 2);  // NCHW padded
    unsigned short* Y0p    = (unsigned short*)alloc(PADN * 2);  // NHWC padded
    unsigned short* Y1p    = (unsigned short*)alloc(PADN * 2);  // NCHW padded
    float*          OUT0   = (float*)alloc(P * 192 * 4);        // NCHW fp32
    float*          g0     = (float*)alloc(1536 * 4);
    float*          gate   = (float*)alloc(1536 * 4);
    unsigned short* W0s  = (unsigned short*)alloc(36864 * 2);
    unsigned short* W1s  = (unsigned short*)alloc(36864 * 2);
    unsigned short* Ws0a = (unsigned short*)alloc(36864 * 2);
    unsigned short* Ws0b = (unsigned short*)alloc(36864 * 2);
    unsigned short* Wos  = (unsigned short*)alloc(36864 * 2);
    unsigned short* Ws1s = (unsigned short*)alloc(9 * 36864 * 2);

    // Zero padded activation buffers (borders must be 0 each call).
    fill_zero_kernel<<<2048, 256, 0, stream>>>((unsigned int*)X2p, PADN / 2);
    fill_zero_kernel<<<2048, 256, 0, stream>>>((unsigned int*)Y0p, PADN / 2);
    fill_zero_kernel<<<2048, 256, 0, stream>>>((unsigned int*)Y1p, PADN / 2);

    // Weight swizzle (bf16, B-fragment layout).
    swizzleB_kernel<<<144, 256, 0, stream>>>(w0, 192, 0, W0s);
    swizzleB_kernel<<<144, 256, 0, stream>>>(w1, 192, 0, W1s);
    swizzleB_kernel<<<144, 256, 0, stream>>>(ws0, 384, 0, Ws0a);    // x part
    swizzleB_kernel<<<144, 256, 0, stream>>>(ws0, 384, 192, Ws0b);  // centers part
    swizzleB_kernel<<<144, 256, 0, stream>>>(wo, 192, 0, Wos);
    swizzleB_conv3_kernel<<<1296, 256, 0, stream>>>(ws1, Ws1s);

    // Activations: NCHW->NHWC bf16 + mask-gathered centers.
    prep_acts_kernel<<<24576, 256, 0, stream>>>(input, centers, mask, X0bf, CEbf);

    // x = conv1(lrelu(conv0(input)))
    gemm192_kernel<6><<<256, 256, 0, stream>>>(X0bf, nullptr, W0s, nullptr, b0, 1,
                                               X1bf, nullptr, nullptr);
    gemm192_kernel<6><<<256, 256, 0, stream>>>(X1bf, nullptr, W1s, nullptr, b1, 0,
                                               X2bf, nullptr, X2p);
    // y = lrelu(ws0 @ concat(x, centers))  -> padded NHWC for the 3x3 conv
    gemm192_kernel<12><<<256, 256, 0, stream>>>(X2bf, CEbf, Ws0a, Ws0b, bs0, 1,
                                                nullptr, Y0p, nullptr);
    // y = lrelu(conv3x3(y)) -> padded NCHW; LDS-staged halo tile (101 KB/WG)
    conv3x3_kernel<<<256, 256, 4 * 66 * C3_PITCH * 2, stream>>>(Y0p, Ws1s, bs1, Y1p);
    // fused k_sp generation + dynamic conv -> OUT0 (fp32 NCHW)
    kdyn_kernel<<<1536, 256, 0, stream>>>(Y1p, X2p, ws2, bs2, OUT0);
    // squeeze-excite
    gap_reduce_kernel<<<1536, 256, 0, stream>>>(OUT0, g0);
    se_gate_kernel<<<8, 192, 0, stream>>>(g0, wc0, bc0, wc1, bc1, gate);
    gate_apply_kernel<<<24576, 256, 0, stream>>>(OUT0, gate, OUT1bf);
    // conv_out + residual
    gemm_out_kernel<<<256, 256, 0, stream>>>(OUT1bf, Wos, bo, input, out);
}
